// Trans_CA_50749333569859
// MI455X (gfx1250) — matrix operand order repaired and checked by reading, not compiled, not run on hardware
//
#include <hip/hip_runtime.h>
#include <hip/hip_bf16.h>

// MI455X / gfx1250 implementation.
// Compute-bound on the two 3x3 convs (~109 GFLOP); routed through
// v_wmma_f32_16x16x32_bf16 implicit GEMM with LDS-staged NHWC-bf16 activations
// and pre-packed A-fragment weights. Attention tail uses the same WMMA GEMM for
// k/v/scores; softmax + gate + final elementwise scale are bandwidth-trivial.

#define DIM 192

typedef __attribute__((ext_vector_type(16))) __bf16 v16bf;
typedef __attribute__((ext_vector_type(8)))  float  v8f;
typedef __attribute__((ext_vector_type(8)))  unsigned short ushort8;

__device__ __forceinline__ unsigned short f2bf(float f) {
    union { float f; unsigned u; } v; v.f = f;
    unsigned u = v.u;
    u += 0x7fffu + ((u >> 16) & 1u);   // round-to-nearest-even
    return (unsigned short)(u >> 16);
}
__device__ __forceinline__ float bf2f(unsigned short h) {
    union { unsigned u; float f; } v; v.u = ((unsigned)h) << 16;
    return v.f;
}

// ---------------------------------------------------------------------------
// Pack a [192 x 192 x ntaps] weight (OIHW conv or O x I matrix) into WMMA
// A-fragment order: frag f = (mt*ntaps + t)*6 + kb ; per lane 16 contiguous
// bf16, k = 16*(lane/16) + j (contiguous order; B operand uses the same).
// ---------------------------------------------------------------------------
__global__ void prepack_w(const float* __restrict__ w, unsigned short* __restrict__ dst,
                          int ntaps, int tapstride) {
    int tid = blockIdx.x * blockDim.x + threadIdx.x;
    int total = 12 * ntaps * 6 * 512;
    if (tid >= total) return;
    int f = tid >> 9;
    int r = tid & 511;
    int lane = r >> 4, j = r & 15;
    int kb = f % 6;
    int t  = (f / 6) % ntaps;
    int mt = f / (6 * ntaps);
    int laneHi = lane >> 4;                       // 0 or 1
    int Mloc = lane & 15;
    int Kloc = j + laneHi * 16;                   // {0..15} / {16..31}
    int ic = kb * 32 + Kloc;
    int M  = mt * 16 + Mloc;
    dst[tid] = f2bf(w[((size_t)M * DIM + ic) * tapstride + t]);
}

// ---------------------------------------------------------------------------
// NCHW fp32 -> NHWC bf16 (channel-contiguous) via LDS tile transpose.
// grid: (W/32, H, B), block 256
// ---------------------------------------------------------------------------
__global__ void nchw_to_nhwc_bf16(const float* __restrict__ x, unsigned short* __restrict__ out,
                                  int H, int W) {
    __shared__ float tile[32][33];
    const int b = blockIdx.z, h = blockIdx.y, w0 = blockIdx.x * 32;
    const int tid = threadIdx.x;
    const int tlo = tid & 31, thi = tid >> 5;     // 0..31 , 0..7
    for (int cb = 0; cb < 6; ++cb) {
        int c0 = cb * 32;
        for (int k = 0; k < 4; ++k) {
            int ci = thi + k * 8;
            tile[ci][tlo] = x[(((size_t)b * DIM + c0 + ci) * H + h) * W + w0 + tlo];
        }
        __syncthreads();
        for (int k = 0; k < 4; ++k) {
            int wj = thi + k * 8;
            out[(((size_t)b * H + h) * W + w0 + wj) * DIM + c0 + tlo] = f2bf(tile[tlo][wj]);
        }
        __syncthreads();
    }
}

// ---------------------------------------------------------------------------
// Fused 3x3 conv (SAME) + bias + ReLU + 2x2 maxpool, implicit GEMM via WMMA.
// in  : NHWC bf16 [B,H,W,192]      out: NHWC bf16 [B,H/2,W/2,192]
// grid: ((W/16)*(H/2), B), block 384 (12 waves; wave = 16-channel M tile).
// Each wave keeps two accumulators (output rows h0, h0+1) so pooling is local.
// ---------------------------------------------------------------------------
__global__ void __launch_bounds__(384)
conv_pool_wmma(const unsigned short* __restrict__ in, const unsigned short* __restrict__ wfrag,
               const float* __restrict__ bias, unsigned short* __restrict__ out,
               int H, int W) {
    __shared__ __align__(32) unsigned short sPatch[4 * 18 * DIM];  // 27.6 KB halo patch
    const int b = blockIdx.y;
    const int wblocks = W >> 4;
    const int h0 = (blockIdx.x / wblocks) * 2;
    const int w0 = (blockIdx.x % wblocks) * 16;
    const int tid = threadIdx.x;

    // Stage 4 rows x 18 cols x 192 ch (zero-padded halo), channel-contiguous.
    for (int i = tid; i < 4 * 18 * DIM; i += 384) {
        int r   = i / (18 * DIM);
        int rem = i - r * 18 * DIM;
        int c   = rem / DIM;
        int ic  = rem - c * DIM;
        int gh = h0 - 1 + r, gw = w0 - 1 + c;
        unsigned short v = 0;
        if (gh >= 0 && gh < H && gw >= 0 && gw < W)
            v = in[(((size_t)b * H + gh) * W + gw) * DIM + ic];
        sPatch[i] = v;
    }
    __syncthreads();

    const int wv    = tid >> 5;     // M tile 0..11
    const int lane  = tid & 31;
    const int col   = lane & 15;    // pixel within 16-wide row (B-matrix N)
    const int khalf = lane >> 4;    // 0: K 0..15 run, 1: K 16..31 run

    v8f acc0 = {}; v8f acc1 = {};
    const unsigned short* abase = wfrag + ((size_t)wv * 9 * 6) * 512 + lane * 16;

    for (int tap = 0; tap < 9; ++tap) {
        const int dh = tap / 3, dw = tap - dh * 3;
        const unsigned short* p0 = &sPatch[((0 + dh) * 18 + (col + dw)) * DIM + khalf * 16];
        const unsigned short* p1 = &sPatch[((1 + dh) * 18 + (col + dw)) * DIM + khalf * 16];
        for (int kb = 0; kb < 6; ++kb) {
            v16bf a  = *(const v16bf*)(abase + (size_t)(tap * 6 + kb) * 512);
            v16bf b0 = *(const v16bf*)(p0 + kb * 32);
            v16bf b1 = *(const v16bf*)(p1 + kb * 32);
            acc0 = __builtin_amdgcn_wmma_f32_16x16x32_bf16(false, a, false, b0,
                                                           (short)0, acc0, false, false);
            acc1 = __builtin_amdgcn_wmma_f32_16x16x32_bf16(false, a, false, b1,
                                                           (short)0, acc1, false, false);
        }
    }

    // 2x2 maxpool: rows via acc0/acc1, cols via lane ^ 1; then bias + ReLU.
    const int Ho = H >> 1, Wo = W >> 1;
    const int chBase = wv * 16 + khalf * 8;
    ushort8 pk;
#pragma unroll
    for (int r = 0; r < 8; ++r) {
        float t = fmaxf(acc0[r], acc1[r]);
        float o = __shfl_xor(t, 1, 32);
        float p = fmaxf(t, o);
        p = fmaxf(p + bias[chBase + r], 0.0f);
        pk[r] = f2bf(p);
    }
    if ((col & 1) == 0) {
        int c = col >> 1;
        size_t base = ((((size_t)b * Ho + (h0 >> 1)) * Wo) + (w0 >> 1) + c) * DIM + chBase;
        *(ushort8*)(&out[base]) = pk;
    }
}

// ---------------------------------------------------------------------------
// out[b][n][m] = sum_c A[m][c] * in[b][n][c]   (A = prepacked 192x192 frags)
// in: bf16 [B,N,192]; out bf16 or fp32 [B,N,192].  grid (N/32, B), block 384.
// ---------------------------------------------------------------------------
__global__ void __launch_bounds__(384)
gemm192_wmma(const unsigned short* __restrict__ in, const unsigned short* __restrict__ wfrag,
             void* __restrict__ out, int N, int out_bf16) {
    __shared__ __align__(32) unsigned short sTok[32 * DIM];   // 12.3 KB
    const int b = blockIdx.y;
    const int n0 = blockIdx.x * 32;
    const int tid = threadIdx.x;
    for (int i = tid; i < 32 * DIM; i += 384)
        sTok[i] = in[((size_t)b * N + n0) * DIM + i];
    __syncthreads();

    const int wv = tid >> 5, lane = tid & 31;
    const int col = lane & 15, khalf = lane >> 4;
    v8f acc0 = {}; v8f acc1 = {};
    const unsigned short* abase = wfrag + (size_t)wv * 6 * 512 + lane * 16;
    const unsigned short* p0 = &sTok[(size_t)col * DIM + khalf * 16];
    const unsigned short* p1 = &sTok[(size_t)(col + 16) * DIM + khalf * 16];
    for (int kb = 0; kb < 6; ++kb) {
        v16bf a  = *(const v16bf*)(abase + (size_t)kb * 512);
        v16bf b0 = *(const v16bf*)(p0 + kb * 32);
        v16bf b1 = *(const v16bf*)(p1 + kb * 32);
        acc0 = __builtin_amdgcn_wmma_f32_16x16x32_bf16(false, a, false, b0,
                                                       (short)0, acc0, false, false);
        acc1 = __builtin_amdgcn_wmma_f32_16x16x32_bf16(false, a, false, b1,
                                                       (short)0, acc1, false, false);
    }
    const int chBase = wv * 16 + khalf * 8;
    size_t i0 = (((size_t)b * N + n0 + col) * DIM) + chBase;
    size_t i1 = (((size_t)b * N + n0 + 16 + col) * DIM) + chBase;
    if (out_bf16) {
        unsigned short* ob = (unsigned short*)out;
        ushort8 q0, q1;
#pragma unroll
        for (int r = 0; r < 8; ++r) { q0[r] = f2bf(acc0[r]); q1[r] = f2bf(acc1[r]); }
        *(ushort8*)(&ob[i0]) = q0;
        *(ushort8*)(&ob[i1]) = q1;
    } else {
        float* of = (float*)out;
        *(float4*)(&of[i0])     = make_float4(acc0[0], acc0[1], acc0[2], acc0[3]);
        *(float4*)(&of[i0 + 4]) = make_float4(acc0[4], acc0[5], acc0[6], acc0[7]);
        *(float4*)(&of[i1])     = make_float4(acc1[0], acc1[1], acc1[2], acc1[3]);
        *(float4*)(&of[i1 + 4]) = make_float4(acc1[4], acc1[5], acc1[6], acc1[7]);
    }
}

// ---------------------------------------------------------------------------
// In-place softmax over s (1024) of S[b][s][qi].  grid (192, B), block 256.
// ---------------------------------------------------------------------------
__global__ void softmax_s(float* __restrict__ S) {
    __shared__ float red[256];
    const int qi = blockIdx.x, b = blockIdx.y, tid = threadIdx.x;
    float vals[4]; float m = -3.0e38f;
    for (int k = 0; k < 4; ++k) {
        int s = tid + k * 256;
        vals[k] = S[(((size_t)b * 1024 + s) * DIM) + qi];
        m = fmaxf(m, vals[k]);
    }
    red[tid] = m; __syncthreads();
    for (int st = 128; st > 0; st >>= 1) {
        if (tid < st) red[tid] = fmaxf(red[tid], red[tid + st]);
        __syncthreads();
    }
    m = red[0]; __syncthreads();
    float sum = 0.f;
    for (int k = 0; k < 4; ++k) { vals[k] = __expf(vals[k] - m); sum += vals[k]; }
    red[tid] = sum; __syncthreads();
    for (int st = 128; st > 0; st >>= 1) {
        if (tid < st) red[tid] += red[tid + st];
        __syncthreads();
    }
    float inv = 1.0f / red[0];
    for (int k = 0; k < 4; ++k)
        S[(((size_t)b * 1024 + tid + k * 256) * DIM) + qi] = vals[k] * inv;
}

// ---------------------------------------------------------------------------
// a[b,qi,d] = sum_s attn[b,s,qi] * v[b,s,d]; gate = relu(a . dim_w[qi] + b)+1
// grid (192, B), block 256 (d = tid < 192 active).
// ---------------------------------------------------------------------------
__global__ void attn_gate(const float* __restrict__ attn, const unsigned short* __restrict__ v,
                          const float* __restrict__ dim_w, const float* __restrict__ dim_b,
                          float* __restrict__ gate) {
    __shared__ float sA[DIM];
    __shared__ float red[256];
    const int qi = blockIdx.x, b = blockIdx.y, tid = threadIdx.x;
    const int d = tid;
    float acc = 0.f;
    for (int s0 = 0; s0 < 1024; s0 += DIM) {
        int sc = (1024 - s0 < DIM) ? (1024 - s0) : DIM;
        if (tid < sc) sA[tid] = attn[(((size_t)b * 1024 + s0 + tid) * DIM) + qi];
        __syncthreads();
        if (d < DIM) {
            for (int ss = 0; ss < sc; ++ss)
                acc += sA[ss] * bf2f(v[(((size_t)b * 1024 + s0 + ss) * DIM) + d]);
        }
        __syncthreads();
    }
    red[tid] = (d < DIM) ? acc * dim_w[(size_t)qi * DIM + d] : 0.f;
    __syncthreads();
    for (int st = 128; st > 0; st >>= 1) {
        if (tid < st) red[tid] += red[tid + st];
        __syncthreads();
    }
    if (tid == 0) gate[b * DIM + qi] = fmaxf(red[0] + dim_b[qi], 0.f) + 1.0f;
}

// ---------------------------------------------------------------------------
// out[b,c,h,w] = x[b,c,h,w] * gate[b,c]   (streaming, float4)
// ---------------------------------------------------------------------------
__global__ void scale_out(const float4* __restrict__ x4, const float* __restrict__ gate,
                          float4* __restrict__ o4, int n4) {
    for (size_t i = (size_t)blockIdx.x * blockDim.x + threadIdx.x; i < (size_t)n4;
         i += (size_t)gridDim.x * blockDim.x) {
        size_t f = i * 4;
        int bc = (int)(f >> 14);            // / (128*128)
        float g = gate[(bc / DIM) * DIM + (bc % DIM)];
        float4 v = x4[i];
        v.x *= g; v.y *= g; v.z *= g; v.w *= g;
        o4[i] = v;
    }
}

// ---------------------------------------------------------------------------
extern "C" void kernel_launch(void* const* d_in, const int* in_sizes, int n_in,
                              void* d_out, int out_size, void* d_ws, size_t ws_size,
                              hipStream_t stream) {
    (void)in_sizes; (void)n_in; (void)out_size; (void)ws_size;
    const float* x     = (const float*)d_in[0];
    const float* c1w   = (const float*)d_in[1];
    const float* c1b   = (const float*)d_in[2];
    const float* c2w   = (const float*)d_in[3];
    const float* c2b   = (const float*)d_in[4];
    const float* query = (const float*)d_in[5];
    const float* keyw  = (const float*)d_in[6];
    const float* valw  = (const float*)d_in[7];
    const float* dimw  = (const float*)d_in[8];
    const float* dimb  = (const float*)d_in[9];
    float* out = (float*)d_out;

    char* ws = (char*)d_ws;
    size_t off = 0;
    auto alloc = [&](size_t bytes) -> void* {
        void* p = ws + off;
        off = (off + bytes + 255) & ~(size_t)255;
        return p;
    };
    const size_t CONV_FRAG  = (size_t)12 * 9 * 6 * 512 * 2;  // 663552 B
    const size_t MAT_FRAG   = (size_t)12 * 1 * 6 * 512 * 2;  //  73728 B
    unsigned short* A1  = (unsigned short*)alloc(CONV_FRAG);
    unsigned short* A2  = (unsigned short*)alloc(CONV_FRAG);
    unsigned short* AK  = (unsigned short*)alloc(MAT_FRAG);
    unsigned short* AV  = (unsigned short*)alloc(MAT_FRAG);
    unsigned short* AQ  = (unsigned short*)alloc(MAT_FRAG);
    unsigned short* XH  = (unsigned short*)alloc((size_t)8 * 128 * 128 * DIM * 2);
    unsigned short* X1  = (unsigned short*)alloc((size_t)8 * 64 * 64 * DIM * 2);
    unsigned short* TOK = (unsigned short*)alloc((size_t)8 * 1024 * DIM * 2);
    unsigned short* KB  = (unsigned short*)alloc((size_t)8 * 1024 * DIM * 2);
    unsigned short* VB  = (unsigned short*)alloc((size_t)8 * 1024 * DIM * 2);
    float*          S   = (float*)alloc((size_t)8 * 1024 * DIM * 4);
    float*          G   = (float*)alloc((size_t)8 * DIM * 4);

    // weight prepack
    hipLaunchKernelGGL(prepack_w, dim3((12 * 9 * 6 * 512 + 255) / 256), dim3(256), 0, stream,
                       c1w, A1, 9, 9);
    hipLaunchKernelGGL(prepack_w, dim3((12 * 9 * 6 * 512 + 255) / 256), dim3(256), 0, stream,
                       c2w, A2, 9, 9);
    hipLaunchKernelGGL(prepack_w, dim3((12 * 6 * 512 + 255) / 256), dim3(256), 0, stream,
                       keyw, AK, 1, 1);
    hipLaunchKernelGGL(prepack_w, dim3((12 * 6 * 512 + 255) / 256), dim3(256), 0, stream,
                       valw, AV, 1, 1);
    hipLaunchKernelGGL(prepack_w, dim3((12 * 6 * 512 + 255) / 256), dim3(256), 0, stream,
                       query, AQ, 1, 1);

    // layout conversion + conv stem
    hipLaunchKernelGGL(nchw_to_nhwc_bf16, dim3(128 / 32, 128, 8), dim3(256), 0, stream,
                       x, XH, 128, 128);
    hipLaunchKernelGGL(conv_pool_wmma, dim3((128 / 16) * (128 / 2), 8), dim3(384), 0, stream,
                       XH, A1, c1b, X1, 128, 128);
    hipLaunchKernelGGL(conv_pool_wmma, dim3((64 / 16) * (64 / 2), 8), dim3(384), 0, stream,
                       X1, A2, c2b, TOK, 64, 64);

    // attention tail
    hipLaunchKernelGGL(gemm192_wmma, dim3(1024 / 32, 8), dim3(384), 0, stream,
                       TOK, AK, (void*)KB, 1024, 1);                  // k  (bf16)
    hipLaunchKernelGGL(gemm192_wmma, dim3(1024 / 32, 8), dim3(384), 0, stream,
                       TOK, AV, (void*)VB, 1024, 1);                  // v  (bf16)
    hipLaunchKernelGGL(gemm192_wmma, dim3(1024 / 32, 8), dim3(384), 0, stream,
                       KB, AQ, (void*)S, 1024, 0);                    // scores (f32)
    hipLaunchKernelGGL(softmax_s, dim3(DIM, 8), dim3(256), 0, stream, S);
    hipLaunchKernelGGL(attn_gate, dim3(DIM, 8), dim3(256), 0, stream,
                       S, VB, dimw, dimb, G);

    // final elementwise scale
    const int n4 = 8 * DIM * 128 * 128 / 4;
    hipLaunchKernelGGL(scale_out, dim3(4096), dim3(256), 0, stream,
                       (const float4*)x, G, (float4*)out, n4);
}
